// ContextQueryAtt_3770981286197
// MI455X (gfx1250) — compile-verified, matrix operand order checked by reading, not run on hardware
//
#include <hip/hip_runtime.h>

#define NEGV (-1e30f)
#define FNEG (-3.0e38f)

static constexpr int BS = 32, CL = 1024, QL = 128, D = 512;

typedef __attribute__((ext_vector_type(16))) __bf16 v16bf;
typedef __attribute__((ext_vector_type(8)))  __bf16 v8bf;
typedef __attribute__((ext_vector_type(8)))  float  v8f;

// ---------------- workspace layout (all chunks 256B-aligned) ----------------
static constexpr size_t SZ_CTXB = (size_t)BS * CL * D * 2;   // context bf16
static constexpr size_t SZ_QWB  = (size_t)BS * QL * D * 2;   // (query*Wcq) bf16
static constexpr size_t SZ_QB   = (size_t)BS * QL * D * 2;   // query bf16
static constexpr size_t SZ_CSIM = (size_t)BS * CL * 4;       // context.Wc
static constexpr size_t SZ_QSIM = (size_t)BS * QL * 4;       // query.Wq
static constexpr size_t SZ_SIM  = (size_t)BS * CL * QL * 4;  // raw sim f32
static constexpr size_t SZ_S1B  = (size_t)BS * CL * QL * 2;  // S1 bf16
static constexpr size_t SZ_PST  = (size_t)BS * 8 * QL * 4;   // partial col stats
static constexpr size_t SZ_CST  = (size_t)BS * QL * 4;       // col max / col sum
static constexpr size_t SZ_TB   = (size_t)BS * QL * D * 2;   // T = S2^T C, bf16

static constexpr size_t OFF_CTXB = 0;
static constexpr size_t OFF_QWB  = OFF_CTXB + SZ_CTXB;
static constexpr size_t OFF_QB   = OFF_QWB  + SZ_QWB;
static constexpr size_t OFF_CSIM = OFF_QB   + SZ_QB;
static constexpr size_t OFF_QSIM = OFF_CSIM + SZ_CSIM;
static constexpr size_t OFF_SIM  = OFF_QSIM + SZ_QSIM;
static constexpr size_t OFF_S1B  = OFF_SIM  + SZ_SIM;
static constexpr size_t OFF_PMAX = OFF_S1B  + SZ_S1B;
static constexpr size_t OFF_PSUM = OFF_PMAX + SZ_PST;
static constexpr size_t OFF_CMAX = OFF_PSUM + SZ_PST;
static constexpr size_t OFF_CSUM = OFF_CMAX + SZ_CST;
static constexpr size_t OFF_TB   = OFF_CSUM + SZ_CST;

// ---------------- helpers ----------------
__device__ inline v8f zero8() {
  v8f z;
#pragma unroll
  for (int i = 0; i < 8; ++i) z[i] = 0.0f;
  return z;
}

__device__ inline v8f wmma_bf16(v16bf a, v16bf b, v8f c) {
  // D = A(16x32 bf16) * B(32x16 bf16) + C(16x16 f32)
  return __builtin_amdgcn_wmma_f32_16x16x32_bf16(false, a, false, b,
                                                 (short)0, c, false, false);
}

// Load a 16-element bf16 WMMA fragment for a matrix stored with the fragment's
// "row" index (M for A, N for B) as the slow dim and K contiguous.
// ISA layout: lane holds row = row0 + (lane&15); K = (j<8?0:16) + (lane>>4)*8 + j%8
// => two contiguous 8xbf16 (16B) loads per lane.
__device__ inline v16bf frag_ld_rowmajor(const __bf16* base, int ld,
                                         int row0, int k0, int lane) {
  const int r = lane & 15, half = lane >> 4;
  const __bf16* p = base + (size_t)(row0 + r) * ld + k0 + half * 8;
  union { v16bf v; v8bf h[2]; } u;
  u.h[0] = *(const v8bf*)(p);
  u.h[1] = *(const v8bf*)(p + 16);
  return u.v;
}

// reductions across a 16-lane half-wave (xor masks <=8 keep bit4 = half id)
__device__ inline float hmax16(float x) {
#pragma unroll
  for (int m = 8; m >= 1; m >>= 1) x = fmaxf(x, __shfl_xor(x, m, 32));
  return x;
}
__device__ inline float hsum16(float x) {
#pragma unroll
  for (int m = 8; m >= 1; m >>= 1) x += __shfl_xor(x, m, 32);
  return x;
}

// ---------------- K0: bf16 casts + row dot with weight vector ----------------
// one 128-thread block per row of length D
__global__ void k_prep(const float* __restrict__ src, const float* __restrict__ wvec,
                       const float* __restrict__ wcq, __bf16* __restrict__ dstb,
                       __bf16* __restrict__ dst_scaled, float* __restrict__ rowdot) {
  const int row = blockIdx.x;
  const float* s = src + (size_t)row * D;
  __shared__ float red[128];
  float part = 0.0f;
  for (int d = threadIdx.x; d < D; d += 128) {
    float v = s[d];
    dstb[(size_t)row * D + d] = (__bf16)v;
    if (dst_scaled) dst_scaled[(size_t)row * D + d] = (__bf16)(v * wcq[d]);
    part += v * wvec[d];
  }
  red[threadIdx.x] = part;
  __syncthreads();
  for (int s2 = 64; s2 > 0; s2 >>= 1) {
    if (threadIdx.x < s2) red[threadIdx.x] += red[threadIdx.x + s2];
    __syncthreads();
  }
  if (threadIdx.x == 0) rowdot[row] = red[0];
}

// ---------------- K1: sim = C.(Q*Wcq)^T + csim + qsim + bias; row softmax -> S1
// grid (8 c-tiles, BS), 256 threads = 8 waves; wave = 16 c-rows x all 128 q
__global__ void __launch_bounds__(256) k_sim(
    const __bf16* __restrict__ ctxb, const __bf16* __restrict__ qwb,
    const float* __restrict__ csim, const float* __restrict__ qsim,
    const float* __restrict__ qmask, const float* __restrict__ bias,
    float* __restrict__ sim, __bf16* __restrict__ s1b) {
  const int b = blockIdx.y;
  const int c0 = blockIdx.x * 128;
  const int wave = threadIdx.x >> 5, lane = threadIdx.x & 31;
  const int nl = lane & 15, half = lane >> 4;
  const int m0 = c0 + wave * 16;
  const __bf16* Am = ctxb + (size_t)b * CL * D;
  const __bf16* Bm = qwb + (size_t)b * QL * D;

  v8f acc[8];
#pragma unroll
  for (int t = 0; t < 8; ++t) acc[t] = zero8();

  for (int k0 = 0; k0 < D; k0 += 32) {
    v16bf a = frag_ld_rowmajor(Am, D, m0, k0, lane);
#pragma unroll
    for (int t = 0; t < 8; ++t) {
      v16bf bb = frag_ld_rowmajor(Bm, D, t * 16, k0, lane);  // rows = q (N)
      acc[t] = wmma_bf16(a, bb, acc[t]);
    }
  }

  const float bias0 = bias[0];
  float csv[8], qsv[8], qmv[8];
#pragma unroll
  for (int r = 0; r < 8; ++r) csv[r] = csim[(size_t)b * CL + m0 + r + 8 * half];
#pragma unroll
  for (int t = 0; t < 8; ++t) {
    int q = t * 16 + nl;
    qsv[t] = qsim[(size_t)b * QL + q];
    qmv[t] = qmask[(size_t)b * QL + q];
  }

#pragma unroll
  for (int t = 0; t < 8; ++t)
#pragma unroll
    for (int r = 0; r < 8; ++r) {
      float sv = acc[t][r] + csv[r] + qsv[t] + bias0;
      sim[((size_t)b * CL + m0 + r + 8 * half) * QL + t * 16 + nl] = sv;
      acc[t][r] = sv * qmv[t] + (1.0f - qmv[t]) * NEGV;  // q_logits
    }

  // row softmax over q: row m = m0 + r + 8*half lives in one half-wave x 8 tiles
#pragma unroll
  for (int r = 0; r < 8; ++r) {
    float mx = FNEG;
#pragma unroll
    for (int t = 0; t < 8; ++t) mx = fmaxf(mx, acc[t][r]);
    mx = hmax16(mx);
    float sm = 0.0f;
#pragma unroll
    for (int t = 0; t < 8; ++t) {
      float e = __expf(acc[t][r] - mx);
      acc[t][r] = e;
      sm += e;
    }
    sm = hsum16(sm);
    float inv = 1.0f / sm;
#pragma unroll
    for (int t = 0; t < 8; ++t)
      s1b[((size_t)b * CL + m0 + r + 8 * half) * QL + t * 16 + nl] =
          (__bf16)(acc[t][r] * inv);
  }
}

// ---------------- K2a: per-(b, 128-row chunk) column max / exp-sum ------------
__global__ void k_colstat1(const float* __restrict__ sim, const float* __restrict__ cmask,
                           float* __restrict__ pmax, float* __restrict__ psum) {
  const int b = blockIdx.y, chunk = blockIdx.x, q = threadIdx.x;
  const float* simb = sim + (size_t)b * CL * QL;
  const float* cm = cmask + (size_t)b * CL;
  float mx = FNEG;
  for (int c = chunk * 128; c < chunk * 128 + 128; ++c) {
    float m = cm[c];
    float cl = simb[(size_t)c * QL + q] * m + (1.0f - m) * NEGV;
    mx = fmaxf(mx, cl);
  }
  float sm = 0.0f;
  for (int c = chunk * 128; c < chunk * 128 + 128; ++c) {
    float m = cm[c];
    float cl = simb[(size_t)c * QL + q] * m + (1.0f - m) * NEGV;
    sm += __expf(cl - mx);
  }
  pmax[((size_t)b * 8 + chunk) * QL + q] = mx;
  psum[((size_t)b * 8 + chunk) * QL + q] = sm;
}

// ---------------- K2b: combine chunk stats --------------------------------
__global__ void k_colstat2(const float* __restrict__ pmax, const float* __restrict__ psum,
                           float* __restrict__ cmax, float* __restrict__ csum) {
  const int b = blockIdx.x, q = threadIdx.x;
  float gm = FNEG;
#pragma unroll
  for (int ch = 0; ch < 8; ++ch) gm = fmaxf(gm, pmax[((size_t)b * 8 + ch) * QL + q]);
  float gs = 0.0f;
#pragma unroll
  for (int ch = 0; ch < 8; ++ch)
    gs += psum[((size_t)b * 8 + ch) * QL + q] *
          __expf(pmax[((size_t)b * 8 + ch) * QL + q] - gm);
  cmax[(size_t)b * QL + q] = gm;
  csum[(size_t)b * QL + q] = gs;
}

// ---------------- K3: T = S2^T * context  (128 x 512 per batch) -------------
// grid (4 d-tiles, BS), 256 threads; exp(S2 numerator) built in-register,
// context tile transposed through LDS for the B operand.
__global__ void __launch_bounds__(256) k_T(
    const __bf16* __restrict__ ctxb, const float* __restrict__ sim,
    const float* __restrict__ cmask, const float* __restrict__ cmax,
    const float* __restrict__ csum, __bf16* __restrict__ tb) {
  const int b = blockIdx.y;
  const int d0 = blockIdx.x * 128;
  const int tid = threadIdx.x, wave = tid >> 5, lane = tid & 31;
  const int nl = lane & 15, half = lane >> 4;
  const int m0 = wave * 16;  // q rows for this wave
  __shared__ __bf16 ldsB[128 * 32];  // [n][k] transposed context tile
  const __bf16* ctxB = ctxb + (size_t)b * CL * D;
  const float* simb = sim + (size_t)b * CL * QL;
  const float* cm = cmask + (size_t)b * CL;
  const float mxq = cmax[(size_t)b * QL + m0 + nl];  // this lane's A-row shift

  v8f acc[8];
#pragma unroll
  for (int t = 0; t < 8; ++t) acc[t] = zero8();

  for (int k0 = 0; k0 < CL; k0 += 32) {
    __syncthreads();
    for (int idx = tid; idx < 128 * 32; idx += 256) {
      int kk = idx >> 7, nn = idx & 127;  // coalesced along d
      ldsB[nn * 32 + kk] = ctxB[(size_t)(k0 + kk) * D + d0 + nn];
    }
    __syncthreads();
    // A fragment = exp(c_logits - colmax) at (q = m0+nl, c = k0+kk)
    v16bf a;
#pragma unroll
    for (int j = 0; j < 16; ++j) {
      int v = j >> 1, pr = j & 1;
      int kk = ((v < 4) ? 0 : 16) + half * 8 + (v & 3) * 2 + pr;
      int c = k0 + kk;
      float m = cm[c];
      float cl = simb[(size_t)c * QL + m0 + nl] * m + (1.0f - m) * NEGV;
      a[j] = (__bf16)__expf(cl - mxq);
    }
#pragma unroll
    for (int t = 0; t < 8; ++t) {
      v16bf bb = frag_ld_rowmajor(ldsB, 32, t * 16, 0, lane);
      acc[t] = wmma_bf16(a, bb, acc[t]);
    }
  }

  float inv[8];
#pragma unroll
  for (int r = 0; r < 8; ++r)
    inv[r] = 1.0f / csum[(size_t)b * QL + m0 + r + 8 * half];
#pragma unroll
  for (int t = 0; t < 8; ++t)
#pragma unroll
    for (int r = 0; r < 8; ++r)
      tb[((size_t)b * QL + m0 + r + 8 * half) * D + d0 + t * 16 + nl] =
          (__bf16)(acc[t][r] * inv[r]);
}

// ---------------- K4: A = S1*Q, B = S1*T, fused output write ----------------
// grid (4 d-tiles, 8 c-tiles, BS), 256 threads; Q and T tiles transposed
// through one shared 32KB LDS buffer in two phases.
__global__ void __launch_bounds__(256) k_out(
    const float* __restrict__ context, const __bf16* __restrict__ qb,
    const __bf16* __restrict__ tb, const __bf16* __restrict__ s1b,
    float* __restrict__ out) {
  const int d0 = blockIdx.x * 128;
  const int c0 = blockIdx.y * 128;
  const int b = blockIdx.z;
  const int tid = threadIdx.x, wave = tid >> 5, lane = tid & 31;
  const int nl = lane & 15, half = lane >> 4;
  const int m0 = c0 + wave * 16;
  __shared__ __bf16 ldsX[128 * 128];  // 32KB, reused for Q then T
  const __bf16* qB = qb + (size_t)b * QL * D;
  const __bf16* tB = tb + (size_t)b * QL * D;
  const __bf16* s1B = s1b + (size_t)b * CL * QL;

  __builtin_prefetch(context + ((size_t)b * CL + m0) * D + d0, 0, 1);

  // phase 1: stage query tile transposed, accumulate A = S1*Q for all 8 n-tiles
  for (int idx = tid; idx < 128 * 128; idx += 256) {
    int kk = idx >> 7, nn = idx & 127;
    ldsX[nn * 128 + kk] = qB[(size_t)kk * D + d0 + nn];
  }
  __syncthreads();
  v8f accA[8];
#pragma unroll
  for (int t = 0; t < 8; ++t) accA[t] = zero8();
#pragma unroll
  for (int t = 0; t < 8; ++t)
#pragma unroll
    for (int k0 = 0; k0 < QL; k0 += 32) {
      v16bf a = frag_ld_rowmajor(s1B, QL, m0, k0, lane);
      v16bf bq = frag_ld_rowmajor(ldsX, 128, t * 16, k0, lane);
      accA[t] = wmma_bf16(a, bq, accA[t]);
    }
  __syncthreads();

  // phase 2: stage T tile transposed, accumulate B = S1*T, write all 4 segments
  for (int idx = tid; idx < 128 * 128; idx += 256) {
    int kk = idx >> 7, nn = idx & 127;
    ldsX[nn * 128 + kk] = tB[(size_t)kk * D + d0 + nn];
  }
  __syncthreads();
#pragma unroll
  for (int t = 0; t < 8; ++t) {
    v8f accB = zero8();
#pragma unroll
    for (int k0 = 0; k0 < QL; k0 += 32) {
      v16bf a = frag_ld_rowmajor(s1B, QL, m0, k0, lane);
      v16bf bt = frag_ld_rowmajor(ldsX, 128, t * 16, k0, lane);
      accB = wmma_bf16(a, bt, accB);
    }
#pragma unroll
    for (int r = 0; r < 8; ++r) {
      int m = m0 + r + 8 * half;
      int n = d0 + t * 16 + nl;
      float cv = context[((size_t)b * CL + m) * D + n];
      float av = accA[t][r];
      float bv = accB[r];
      float* row = out + ((size_t)b * CL + m) * (size_t)(4 * D);
      row[n] = cv;               // context
      row[D + n] = av;           // A
      row[2 * D + n] = cv * av;  // context * A
      row[3 * D + n] = cv * bv;  // context * B
    }
  }
}

// ---------------- launch ----------------
extern "C" void kernel_launch(void* const* d_in, const int* in_sizes, int n_in,
                              void* d_out, int out_size, void* d_ws, size_t ws_size,
                              hipStream_t stream) {
  const float* context = (const float*)d_in[0];
  const float* query   = (const float*)d_in[1];
  const float* c_mask  = (const float*)d_in[2];
  const float* q_mask  = (const float*)d_in[3];
  const float* Wq      = (const float*)d_in[4];
  const float* Wc      = (const float*)d_in[5];
  const float* Wcq     = (const float*)d_in[6];
  const float* bias    = (const float*)d_in[7];
  float* out = (float*)d_out;

  char* w = (char*)d_ws;
  __bf16* ctxb = (__bf16*)(w + OFF_CTXB);
  __bf16* qwb  = (__bf16*)(w + OFF_QWB);
  __bf16* qb   = (__bf16*)(w + OFF_QB);
  float*  csim = (float*)(w + OFF_CSIM);
  float*  qsim = (float*)(w + OFF_QSIM);
  float*  sim  = (float*)(w + OFF_SIM);
  __bf16* s1b  = (__bf16*)(w + OFF_S1B);
  float*  pmax = (float*)(w + OFF_PMAX);
  float*  psum = (float*)(w + OFF_PSUM);
  float*  cmax = (float*)(w + OFF_CMAX);
  float*  csum = (float*)(w + OFF_CSUM);
  __bf16* tb   = (__bf16*)(w + OFF_TB);

  k_prep<<<dim3(BS * QL), dim3(128), 0, stream>>>(query, Wq, Wcq, qb, qwb, qsim);
  k_prep<<<dim3(BS * CL), dim3(128), 0, stream>>>(context, Wc, nullptr, ctxb, nullptr, csim);
  k_sim<<<dim3(8, BS), dim3(256), 0, stream>>>(ctxb, qwb, csim, qsim, q_mask, bias, sim, s1b);
  k_colstat1<<<dim3(8, BS), dim3(128), 0, stream>>>(sim, c_mask, pmax, psum);
  k_colstat2<<<dim3(BS), dim3(128), 0, stream>>>(pmax, psum, cmax, csum);
  k_T<<<dim3(4, BS), dim3(256), 0, stream>>>(ctxb, sim, c_mask, cmax, csum, tb);
  k_out<<<dim3(4, 8, BS), dim3(256), 0, stream>>>(context, qb, tb, s1b, out);
}
